// _HRETMLP_11897059410390
// MI455X (gfx1250) — matrix operand order repaired and checked by reading, not compiled, not run on hardware
//
#include <hip/hip_runtime.h>
#include <hip/hip_bf16.h>

// ---------------- problem constants ----------------
#define BB    1024
#define NTOK  121
#define DD    1024
#define DHH   675
#define M1    (BB * NTOK)      // 123904 rows (multiple of 128)
#define N0P   1408             // lin0 N padded: 1350 -> 1408 (11 * 128)
#define K1P   704              // lin1 K padded: 675  -> 704  (22 * 32)
#define NHEAD 3072             // gh(512) + r(256) + a(256) + 4*expert(512)

// ---------------- vector types ----------------
typedef __attribute__((ext_vector_type(16))) __bf16 v16bf;
typedef __attribute__((ext_vector_type(8)))  float  v8f;
typedef __attribute__((ext_vector_type(4)))  unsigned int v4u;
typedef __attribute__((ext_vector_type(8)))  int    v8i;
typedef __attribute__((ext_vector_type(4)))  int    v4i;

union FragBF { v16bf v; unsigned int u[8]; };
union FragC  { v8f   v; float        f[8]; };

#if defined(__has_builtin)
#if __has_builtin(__builtin_amdgcn_tensor_load_to_lds) && __has_builtin(__builtin_amdgcn_s_wait_tensorcnt)
#define HAVE_TDM 1
#endif
#endif
#ifndef HAVE_TDM
#define HAVE_TDM 0
#endif

__device__ inline unsigned short f2bf(float f) {
  unsigned int x = __float_as_uint(f);
  unsigned int r = (x + 0x7fffu + ((x >> 16) & 1u)) >> 16;   // RNE
  return (unsigned short)r;
}
__device__ inline float bf2f(unsigned short h) {
  return __uint_as_float(((unsigned int)h) << 16);
}
__device__ inline float gelu_exact(float x) {
  return 0.5f * x * (1.0f + erff(x * 0.70710678118654752f));
}
__device__ inline unsigned lds_off_of(const void* p) {
  // generic pointer to __shared__ : low 32 bits carry the LDS byte offset
  return (unsigned)(unsigned long long)(uintptr_t)p;
}

__device__ inline float block_reduce_sum(float v, float* sbuf) {
  int tid = threadIdx.x;
  sbuf[tid] = v;
  __syncthreads();
  #pragma unroll
  for (int s = 128; s > 0; s >>= 1) {
    if (tid < s) sbuf[tid] += sbuf[tid + s];
    __syncthreads();
  }
  float r = sbuf[0];
  __syncthreads();
  return r;
}

// ---------------- WMMA bf16 GEMM: C = act(A @ W + bias) ----------------
// A: (M x K) bf16 row-major, W: (K x N) bf16 row-major. M%128==0, N%128==0, K%32==0.
// Block = 256 threads = 8 wave32. Block tile 128x128, K-step 32, double-buffered LDS.
// Waves: 4 (M) x 2 (N); per-wave 32x64 = 2x4 WMMA 16x16 tiles.
// A tile staged by the Tensor Data Mover (2D descriptor per K-step, LDS pad
// 64B row + 16B pad == stride 40 bf16) into the *next* buffer while the 8
// WMMAs of the current tile execute; B tile staged transposed by the waves.
#define TILE_K 32
#define LDA_S  40   // LDS A stride (bf16 units): 64B data + 16B pad
#define LDB_S  40   // LDS B^T stride (bf16 units)

__device__ inline void load_a_frag(const unsigned short* ldsA, int rowBase, int lane, FragBF& f) {
  const unsigned int* p32 = (const unsigned int*)ldsA;
  int m  = rowBase + (lane & 15);
  int h4 = ((lane >> 4) & 1) << 2;                 // half * 4 (u32 units)
  // ISA 16-bit A 16x32 layout -> two aligned 16B runs per lane
  uint4 lo = *(const uint4*)(p32 + m * 20 + h4);       // K pairs {0,2,4,6}+8h
  uint4 hi = *(const uint4*)(p32 + m * 20 + 8 + h4);   // K pairs {16,18,20,22}+8h
  f.u[0] = lo.x; f.u[1] = lo.y; f.u[2] = lo.z; f.u[3] = lo.w;
  f.u[4] = hi.x; f.u[5] = hi.y; f.u[6] = hi.z; f.u[7] = hi.w;
}
__device__ inline void load_b_frag(const unsigned short* ldsB, int nBase, int lane, FragBF& f) {
  const unsigned int* p32 = (const unsigned int*)ldsB;
  int n  = nBase + (lane & 15);
  int h8 = ((lane >> 4) & 1) << 2;                 // half * 4 (u32 units)
  // ISA 16-bit layout (same as A) -> two aligned 16B runs per lane (stride 80B)
  uint4 lo = *(const uint4*)(p32 + n * 20 + h8);
  uint4 hi = *(const uint4*)(p32 + n * 20 + 8 + h8);
  f.u[0] = lo.x; f.u[1] = lo.y; f.u[2] = lo.z; f.u[3] = lo.w;
  f.u[4] = hi.x; f.u[5] = hi.y; f.u[6] = hi.z; f.u[7] = hi.w;
}

// ACT: 0=none, 1=gelu, 2=relu ; OUTBF16: 1 -> bf16 out, 0 -> f32 out
template <int ACT, int OUTBF16>
__global__ __launch_bounds__(256) void wmma_gemm_kernel(
    const unsigned short* __restrict__ A, int lda,
    const unsigned short* __restrict__ W, int ldw,
    const float* __restrict__ bias,
    void* __restrict__ Cout, int ldc, int K)
{
  __shared__ __align__(16) unsigned short ldsA[2][128 * LDA_S]; // 2 x 10 KB
  __shared__ __align__(16) unsigned short ldsB[2][128 * LDB_S]; // 2 x 10 KB

  const int tid   = threadIdx.x;
  const int lane  = tid & 31;
  const int wave  = tid >> 5;
  const int waveM = wave >> 1;   // 0..3 -> 32 rows
  const int waveN = wave & 1;    // 0..1 -> 64 cols
  const int m0 = blockIdx.y * 128;
  const int n0 = blockIdx.x * 128;

  FragC c[2][4];
  #pragma unroll
  for (int mi = 0; mi < 2; ++mi)
    #pragma unroll
    for (int ni = 0; ni < 4; ++ni)
      #pragma unroll
      for (int r = 0; r < 8; ++r) c[mi][ni].f[r] = 0.0f;

#if HAVE_TDM
  // Invariant pieces of the A-tile tensor descriptor (2D tile, 8-byte units):
  // row = 32 bf16 = 64B = 8 units; 128 rows; LDS pad 16B per 64B row.
  const unsigned ldsA_base0 = lds_off_of(ldsA[0]);
  const unsigned ldsA_base1 = lds_off_of(ldsA[1]);
  const unsigned strideU    = (unsigned)(lda >> 2);   // lda*2 bytes / 8
  v8i g1;
  g1[0] = (int)((3u << 16) | (1u << 20) | (3u << 22) | (3u << 25)); // dsz=8B, pad_en, ivl=16DW, amt=4DW
  g1[1] = (int)(8u << 16);                     // tensor_dim0 = 8 units (low16 at bit 48)
  g1[2] = (int)(128u << 16);                   // tensor_dim0 hi=0 | tensor_dim1 = 128 (low16)
  g1[3] = (int)(8u << 16);                     // tensor_dim1 hi=0 | tile_dim0 = 8
  g1[4] = (int)128u;                           // tile_dim1 = 128, tile_dim2 = 0
  g1[5] = (int)strideU;                        // tensor_dim0_stride low 32
  g1[6] = 0;                                   // stride hi16 | tensor_dim1_stride (unused, 2D)
  g1[7] = 0;
  const v4i gz4 = {0, 0, 0, 0};
  const v8i gz8 = {0, 0, 0, 0, 0, 0, 0, 0};
#endif

  // ---- tile staging helpers ----
  auto stageA = [&](int k0, int buf) {
#if HAVE_TDM
    if (wave == 0) {
      unsigned long long ga =
          (unsigned long long)(uintptr_t)(A + (size_t)m0 * lda + k0);
      v4u g0;
      g0[0] = 1u;                                        // count=1 (valid), user mode
      g0[1] = buf ? ldsA_base1 : ldsA_base0;             // D#.lds_addr
      g0[2] = (unsigned)(ga & 0xffffffffu);              // global_addr[31:0]
      g0[3] = (unsigned)((ga >> 32) & 0x01ffffffu) | (2u << 30); // addr[56:32] | type=2
      __builtin_amdgcn_tensor_load_to_lds(g0, g1, gz4, gz4, gz8, 0);
    }
#else
    int ga = tid, gb = tid + 256;
    int ma = ga >> 2, ka = (ga & 3) << 3;
    int mb = gb >> 2, kb = (gb & 3) << 3;
    uint4 va = *(const uint4*)(A + (size_t)(m0 + ma) * lda + k0 + ka);
    uint4 vb = *(const uint4*)(A + (size_t)(m0 + mb) * lda + k0 + kb);
    *(uint4*)(&ldsA[buf][ma * LDA_S + ka]) = va;
    *(uint4*)(&ldsA[buf][mb * LDA_S + kb]) = vb;
#endif
  };
  auto stageB = [&](int k0, int buf) {
    int ga = tid, gb = tid + 256;
    int ka = ga >> 4, nga = (ga & 15) << 3;
    int kb = gb >> 4, ngb = (gb & 15) << 3;
    uint4 va = *(const uint4*)(W + (size_t)(k0 + ka) * ldw + n0 + nga);
    uint4 vb = *(const uint4*)(W + (size_t)(k0 + kb) * ldw + n0 + ngb);
    const unsigned short* sa = (const unsigned short*)&va;
    const unsigned short* sb = (const unsigned short*)&vb;
    #pragma unroll
    for (int i = 0; i < 8; ++i) ldsB[buf][(nga + i) * LDB_S + ka] = sa[i];
    #pragma unroll
    for (int i = 0; i < 8; ++i) ldsB[buf][(ngb + i) * LDB_S + kb] = sb[i];
  };

  // prologue: stage tile 0 into buffer 0
  stageA(0, 0);
  stageB(0, 0);

  int cur = 0;
  for (int k0 = 0; k0 < K; k0 += TILE_K) {
#if HAVE_TDM
    if (wave == 0) __builtin_amdgcn_s_wait_tensorcnt(0);  // A tile of `cur` landed
#endif
    __syncthreads();   // publishes B scatter-stores (dscnt flushed per wave)

    FragBF af[2], bf[4];
    #pragma unroll
    for (int mi = 0; mi < 2; ++mi) load_a_frag(ldsA[cur], waveM * 32 + mi * 16, lane, af[mi]);
    #pragma unroll
    for (int ni = 0; ni < 4; ++ni) load_b_frag(ldsB[cur], waveN * 64 + ni * 16, lane, bf[ni]);

    // kick off staging of the next tile into the other buffer; the 8 WMMAs
    // below overlap the in-flight TDM / global loads.
    int k1 = k0 + TILE_K;
    if (k1 < K) {
      stageA(k1, cur ^ 1);
      stageB(k1, cur ^ 1);
      if (k1 + TILE_K < K) { // warm caches two tiles ahead (global_prefetch_b8)
        __builtin_prefetch(A + (size_t)(m0 + (tid >> 1)) * lda + k1 + TILE_K, 0, 1);
        __builtin_prefetch(W + (size_t)(k1 + TILE_K + lane) * ldw + n0, 0, 1);
      }
    }

    #pragma unroll
    for (int mi = 0; mi < 2; ++mi)
      #pragma unroll
      for (int ni = 0; ni < 4; ++ni)
        c[mi][ni].v = __builtin_amdgcn_wmma_f32_16x16x32_bf16(
            false, af[mi].v, false, bf[ni].v, (short)0, c[mi][ni].v, false, false);

    cur ^= 1;
  }

  // epilogue: bias + activation + store (C layout: M = r + 8*half, N = lane&15)
  const int half = (lane >> 4) & 1;
  const int ncol = lane & 15;
  #pragma unroll
  for (int ni = 0; ni < 4; ++ni) {
    int nn = n0 + waveN * 64 + ni * 16 + ncol;
    float bb = bias ? bias[nn] : 0.0f;
    #pragma unroll
    for (int mi = 0; mi < 2; ++mi) {
      #pragma unroll
      for (int r = 0; r < 8; ++r) {
        int mm = m0 + waveM * 32 + mi * 16 + r + 8 * half;
        float v = c[mi][ni].f[r] + bb;
        if (ACT == 1) v = gelu_exact(v);
        else if (ACT == 2) v = fmaxf(v, 0.0f);
        if (OUTBF16) ((unsigned short*)Cout)[(size_t)mm * ldc + nn] = f2bf(v);
        else         ((float*)Cout)[(size_t)mm * ldc + nn] = v;
      }
    }
  }
}

// ---------------- tokenize + LayerNorm -> a0 (bf16), xr0 (token 0, f32) ----------------
__global__ __launch_bounds__(256) void tokenize_ln_kernel(
    const float* __restrict__ x_num, const int* __restrict__ x_cat,
    const float* __restrict__ tok_w, const float* __restrict__ tok_b,
    const float* __restrict__ cat_emb,
    const float* __restrict__ ln_g, const float* __restrict__ ln_b,
    unsigned short* __restrict__ a0, float* __restrict__ xr0)
{
  __shared__ float sbuf[256];
  const int row = blockIdx.x;        // b*121 + t
  const int b = row / NTOK, t = row % NTOK;
  const int tid = threadIdx.x;
  float vals[4]; float s = 0.f, ss = 0.f;
  #pragma unroll
  for (int i = 0; i < 4; ++i) {
    int d = tid + i * 256;
    float x;
    if (t == 0) {
      x = tok_w[d];                                             // xn=1, bias row 0 = 0
    } else if (t <= 100) {
      x = tok_w[(size_t)t * DD + d] * x_num[b * 100 + (t - 1)] + tok_b[(size_t)(t - 1) * DD + d];
    } else {
      int ci = x_cat[b * 20 + (t - 101)] + (t - 101) * 100;
      x = cat_emb[(size_t)ci * DD + d] + tok_b[(size_t)(t - 1) * DD + d];
    }
    vals[i] = x; s += x; ss += x * x;
  }
  s  = block_reduce_sum(s,  sbuf);
  ss = block_reduce_sum(ss, sbuf);
  float mean = s * (1.0f / DD);
  float var  = ss * (1.0f / DD) - mean * mean;
  float inv  = rsqrtf(var + 1e-5f);
  #pragma unroll
  for (int i = 0; i < 4; ++i) {
    int d = tid + i * 256;
    float xn = (vals[i] - mean) * inv * ln_g[d] + ln_b[d];
    a0[(size_t)row * DD + d] = f2bf(xn);
    if (t == 0) xr0[(size_t)b * DD + d] = vals[i];
  }
}

// ---------------- LN over v = h[:, :, 675:1350] ----------------
__global__ __launch_bounds__(256) void vln_kernel(
    const unsigned short* __restrict__ h,
    const float* __restrict__ g, const float* __restrict__ bta,
    unsigned short* __restrict__ vln)
{
  __shared__ float sbuf[256];
  const int row = blockIdx.x;
  const int tid = threadIdx.x;
  const unsigned short* src = h + (size_t)row * N0P + DHH;  // v columns
  float vals[3]; float s = 0.f, ss = 0.f;
  #pragma unroll
  for (int i = 0; i < 3; ++i) {
    int d = tid + i * 256;
    float x = (d < DHH) ? bf2f(src[d]) : 0.0f;
    vals[i] = x; s += x; ss += x * x;
  }
  s  = block_reduce_sum(s,  sbuf);
  ss = block_reduce_sum(ss, sbuf);
  float mean = s * (1.0f / DHH);
  float var  = ss * (1.0f / DHH) - mean * mean;
  float inv  = rsqrtf(var + 1e-5f);
  #pragma unroll
  for (int i = 0; i < 3; ++i) {
    int d = tid + i * 256;
    if (d < DHH)
      vln[(size_t)row * DHH + d] = f2bf((vals[i] - mean) * inv * g[d] + bta[d]);
  }
}

// ---------------- SGU token-mix (only output token s=0) + u gating ----------------
__global__ __launch_bounds__(256) void sgu_mix_kernel(
    const unsigned short* __restrict__ vln, const unsigned short* __restrict__ h,
    const float* __restrict__ sgu_w, const float* __restrict__ sgu_b,
    unsigned short* __restrict__ uv0)
{
  __shared__ float sw[NTOK];
  const int tid = threadIdx.x;
  if (tid < NTOK) sw[tid] = sgu_w[tid];     // row s=0 of (NT x NT)
  __syncthreads();
  int idx = blockIdx.x * 256 + tid;
  int b = idx / K1P, d = idx % K1P;
  float outv = 0.0f;
  if (d < DHH) {
    float acc = sgu_b[0];
    const unsigned short* vp = vln + (size_t)b * NTOK * DHH + d;
    for (int t = 0; t < NTOK; ++t) acc += sw[t] * bf2f(vp[(size_t)t * DHH]);
    float u = bf2f(h[(size_t)(b * NTOK) * N0P + d]);   // u = token-0 row, cols [0,675)
    outv = u * acc;
  }
  uv0[(size_t)b * K1P + d] = f2bf(outv);               // zero K-padding
}

// ---------------- residual + LN + GELU -> hidden ----------------
__global__ __launch_bounds__(256) void res_ln_gelu_kernel(
    const float* __restrict__ c1, const float* __restrict__ xr0,
    const float* __restrict__ ln_g, const float* __restrict__ ln_b,
    unsigned short* __restrict__ hid_bf, float* __restrict__ hid_f)
{
  __shared__ float sbuf[256];
  const int b = blockIdx.x, tid = threadIdx.x;
  float vals[4]; float s = 0.f, ss = 0.f;
  #pragma unroll
  for (int i = 0; i < 4; ++i) {
    int d = tid + i * 256;
    float x = c1[(size_t)b * DD + d] + xr0[(size_t)b * DD + d];
    vals[i] = x; s += x; ss += x * x;
  }
  s  = block_reduce_sum(s,  sbuf);
  ss = block_reduce_sum(ss, sbuf);
  float mean = s * (1.0f / DD);
  float var  = ss * (1.0f / DD) - mean * mean;
  float inv  = rsqrtf(var + 1e-5f);
  #pragma unroll
  for (int i = 0; i < 4; ++i) {
    int d = tid + i * 256;
    float gv = gelu_exact((vals[i] - mean) * inv * ln_g[d] + ln_b[d]);
    hid_bf[(size_t)b * DD + d] = f2bf(gv);
    hid_f [(size_t)b * DD + d] = gv;
  }
}

// ---------------- weight prep ----------------
__global__ __launch_bounds__(256) void convpad_kernel(
    const float* __restrict__ src, unsigned short* __restrict__ dst,
    int rS, int cS, int rD, int cD)
{
  int idx = blockIdx.x * 256 + threadIdx.x;
  if (idx >= rD * cD) return;
  int r = idx / cD, c = idx % cD;
  float v = (r < rS && c < cS) ? src[(size_t)r * cS + c] : 0.0f;
  dst[idx] = f2bf(v);
}

__global__ __launch_bounds__(256) void whead_kernel(
    const float* __restrict__ gh, const float* __restrict__ rw,
    const float* __restrict__ aw, const float* __restrict__ ex,
    unsigned short* __restrict__ dst)
{
  int idx = blockIdx.x * 256 + threadIdx.x;
  if (idx >= DD * NHEAD) return;
  int k = idx / NHEAD, n = idx % NHEAD;
  float v;
  if      (n <  512) v = gh[(size_t)k * 512 + n];
  else if (n <  768) v = rw[(size_t)k * 256 + (n - 512)];
  else if (n < 1024) v = aw[(size_t)k * 256 + (n - 768)];
  else { int e = (n - 1024) >> 9, m = (n - 1024) & 511;
         v = ex[((size_t)e * DD + k) * 512 + m]; }
  dst[idx] = f2bf(v);
}

__global__ __launch_bounds__(256) void bias_prep_kernel(
    const float* __restrict__ lin0_b,
    const float* __restrict__ ghb, const float* __restrict__ rb,
    const float* __restrict__ ab,  const float* __restrict__ exb,
    float* __restrict__ b0p, float* __restrict__ bh)
{
  int idx = blockIdx.x * 256 + threadIdx.x;
  if (idx < N0P) b0p[idx] = (idx < 1350) ? lin0_b[idx] : 0.0f;
  if (idx < NHEAD) {
    float v;
    if      (idx <  512) v = ghb[idx];
    else if (idx <  768) v = rb[idx - 512];
    else if (idx < 1024) v = ab[idx - 768];
    else                 v = exb[idx - 1024];   // (E,MID) flat
    bh[idx] = v;
  }
}

// ---------------- router / MoE / combine ----------------
__global__ __launch_bounds__(256) void final_kernel(
    const float* __restrict__ ho, const float* __restrict__ hid_f,
    const float* __restrict__ gh_w2, const float* __restrict__ gh_b2,
    const float* __restrict__ r_w2,  const float* __restrict__ r_b2,
    const float* __restrict__ a_w2,  const float* __restrict__ a_b2,
    const float* __restrict__ ex_w2, const float* __restrict__ ex_b2,
    const float* __restrict__ base_w, const float* __restrict__ base_b,
    float* __restrict__ out)
{
  __shared__ float sbuf[256];
  const int b = blockIdx.x, tid = threadIdx.x;
  const float* hr = ho + (size_t)b * NHEAD;

  float glob = 0.f, alpha = 0.f, ybase = 0.f;
  float lg[4] = {0, 0, 0, 0}, eo[4] = {0, 0, 0, 0};
  for (int i = tid; i < 512; i += 256) glob += hr[i] * gh_w2[i];
  {
    int i = tid; // 256 threads cover 256 elems exactly
    float v = hr[512 + i];
    #pragma unroll
    for (int j = 0; j < 4; ++j) lg[j] += v * r_w2[i * 4 + j];
    alpha += hr[768 + i] * a_w2[i];
  }
  #pragma unroll
  for (int e = 0; e < 4; ++e)
    for (int i = tid; i < 512; i += 256) eo[e] += hr[1024 + 512 * e + i] * ex_w2[e * 512 + i];
  for (int i = tid; i < DD; i += 256) ybase += hid_f[(size_t)b * DD + i] * base_w[i];

  glob  = block_reduce_sum(glob,  sbuf);
  #pragma unroll
  for (int j = 0; j < 4; ++j) lg[j] = block_reduce_sum(lg[j], sbuf);
  alpha = block_reduce_sum(alpha, sbuf);
  #pragma unroll
  for (int e = 0; e < 4; ++e) eo[e] = block_reduce_sum(eo[e], sbuf);
  ybase = block_reduce_sum(ybase, sbuf);

  if (tid == 0) {
    glob += gh_b2[0];
    alpha = 1.0f / (1.0f + expf(-(alpha + a_b2[0])));
    ybase += base_b[0];
    float p[4], mx = -1e30f;
    #pragma unroll
    for (int j = 0; j < 4; ++j) { lg[j] += r_b2[j]; mx = fmaxf(mx, lg[j]); }
    float sum = 0.f;
    #pragma unroll
    for (int j = 0; j < 4; ++j) { p[j] = expf(lg[j] - mx); sum += p[j]; }
    #pragma unroll
    for (int j = 0; j < 4; ++j) p[j] /= sum;
    #pragma unroll
    for (int e = 0; e < 4; ++e) eo[e] += ex_b2[e];
    int i1 = 0;
    for (int j = 1; j < 4; ++j) if (p[j] > p[i1]) i1 = j;
    int i2 = (i1 == 0) ? 1 : 0;
    for (int j = 0; j < 4; ++j) if (j != i1 && p[j] > p[i2]) i2 = j;
    float sp = fmaxf(p[i1] + p[i2], 1e-8f);
    float local = (p[i1] * eo[i1] + p[i2] * eo[i2]) / sp;
    out[b] = ybase + alpha * (glob + local);
  }
}

// ---------------- host launch ----------------
extern "C" void kernel_launch(void* const* d_in, const int* in_sizes, int n_in,
                              void* d_out, int out_size, void* d_ws, size_t ws_size,
                              hipStream_t stream) {
  const float* x_num    = (const float*)d_in[0];
  const int*   x_cat    = (const int*)  d_in[1];
  const float* tok_w    = (const float*)d_in[2];
  const float* tok_b    = (const float*)d_in[3];
  const float* cat_emb  = (const float*)d_in[4];
  const float* ln_g     = (const float*)d_in[5];
  const float* ln_b     = (const float*)d_in[6];
  const float* lin0_w   = (const float*)d_in[7];
  const float* lin0_b   = (const float*)d_in[8];
  const float* sgu_ln_g = (const float*)d_in[9];
  const float* sgu_ln_b = (const float*)d_in[10];
  const float* sgu_w    = (const float*)d_in[11];
  const float* sgu_b    = (const float*)d_in[12];
  const float* lin1_w   = (const float*)d_in[13];
  const float* lin1_b   = (const float*)d_in[14];
  const float* r_w1     = (const float*)d_in[15];
  const float* r_b1     = (const float*)d_in[16];
  const float* r_w2     = (const float*)d_in[17];
  const float* r_b2     = (const float*)d_in[18];
  const float* a_w1     = (const float*)d_in[19];
  const float* a_b1     = (const float*)d_in[20];
  const float* a_w2     = (const float*)d_in[21];
  const float* a_b2     = (const float*)d_in[22];
  const float* base_w   = (const float*)d_in[23];
  const float* base_b   = (const float*)d_in[24];
  const float* gh_w1    = (const float*)d_in[25];
  const float* gh_b1    = (const float*)d_in[26];
  const float* gh_w2    = (const float*)d_in[27];
  const float* gh_b2    = (const float*)d_in[28];
  const float* ex_w1    = (const float*)d_in[29];
  const float* ex_b1    = (const float*)d_in[30];
  const float* ex_w2    = (const float*)d_in[31];
  const float* ex_b2    = (const float*)d_in[32];

  char* w = (char*)d_ws;
  size_t off = 0;
  auto take = [&](size_t bytes) -> char* {
    char* p = w + off;
    off = (off + bytes + 255) & ~(size_t)255;
    return p;
  };
  unsigned short* a0    = (unsigned short*)take((size_t)M1 * DD * 2);
  unsigned short* hbuf  = (unsigned short*)take((size_t)M1 * N0P * 2);
  unsigned short* vln   = (unsigned short*)take((size_t)M1 * DHH * 2);
  float*          xr0   = (float*)take((size_t)BB * DD * 4);
  unsigned short* uv0   = (unsigned short*)take((size_t)BB * K1P * 2);
  float*          c1    = (float*)take((size_t)BB * DD * 4);
  unsigned short* hidbf = (unsigned short*)take((size_t)BB * DD * 2);
  float*          hidf  = (float*)take((size_t)BB * DD * 4);
  float*          ho    = (float*)take((size_t)BB * NHEAD * 4);
  unsigned short* w0p   = (unsigned short*)take((size_t)DD * N0P * 2);
  unsigned short* w1p   = (unsigned short*)take((size_t)K1P * DD * 2);
  unsigned short* whp   = (unsigned short*)take((size_t)DD * NHEAD * 2);
  float*          b0p   = (float*)take((size_t)N0P * 4);
  float*          bh    = (float*)take((size_t)NHEAD * 4);
  (void)ws_size; (void)n_in; (void)in_sizes; (void)out_size;

  // --- weight prep (deterministic every call) ---
  convpad_kernel<<<(DD * N0P + 255) / 256, 256, 0, stream>>>(lin0_w, w0p, DD, 1350, DD, N0P);
  convpad_kernel<<<(K1P * DD + 255) / 256, 256, 0, stream>>>(lin1_w, w1p, DHH, DD, K1P, DD);
  whead_kernel<<<(DD * NHEAD + 255) / 256, 256, 0, stream>>>(gh_w1, r_w1, a_w1, ex_w1, whp);
  bias_prep_kernel<<<(NHEAD + 255) / 256, 256, 0, stream>>>(lin0_b, gh_b1, r_b1, a_b1, ex_b1, b0p, bh);

  // --- tokenize + LN ---
  tokenize_ln_kernel<<<M1, 256, 0, stream>>>(x_num, x_cat, tok_w, tok_b, cat_emb, ln_g, ln_b, a0, xr0);

  // --- lin0: (123904 x 1024) @ (1024 x 1408), GELU, bf16 out ---
  wmma_gemm_kernel<1, 1><<<dim3(N0P / 128, M1 / 128), 256, 0, stream>>>(
      a0, DD, w0p, N0P, b0p, (void*)hbuf, N0P, DD);

  // --- LN(v) ---
  vln_kernel<<<M1, 256, 0, stream>>>(hbuf, sgu_ln_g, sgu_ln_b, vln);

  // --- SGU mix (token 0 only) + u gating -> uv0 (B x 704 bf16, zero-padded) ---
  sgu_mix_kernel<<<(BB * K1P) / 256, 256, 0, stream>>>(vln, hbuf, sgu_w, sgu_b, uv0);

  // --- lin1: (1024 x 704) @ (704 x 1024), +bias, f32 out ---
  wmma_gemm_kernel<0, 0><<<dim3(DD / 128, BB / 128), 256, 0, stream>>>(
      uv0, K1P, w1p, DD, lin1_b, (void*)c1, DD, K1P);

  // --- residual + LN + GELU -> hidden ---
  res_ln_gelu_kernel<<<BB, 256, 0, stream>>>(c1, xr0, ln_g, ln_b, hidbf, hidf);

  // --- fused heads GEMM: (1024 x 1024) @ (1024 x 3072), ReLU, f32 out ---
  wmma_gemm_kernel<2, 0><<<dim3(NHEAD / 128, BB / 128), 256, 0, stream>>>(
      hidbf, DD, whp, NHEAD, bh, (void*)ho, NHEAD, DD);

  // --- router / MoE / combine ---
  final_kernel<<<BB, 256, 0, stream>>>(ho, hidf, gh_w2, gh_b2, r_w2, r_b2, a_w2, a_b2,
                                       ex_w2, ex_b2, base_w, base_b, (float*)d_out);
}
